// MoE_52673478918576
// MI455X (gfx1250) — compile-verified, matrix operand order checked by reading, not run on hardware
//
#include <hip/hip_runtime.h>
#include <hip/hip_bf16.h>
#include <math.h>

// ---- problem constants (match reference) ----
#define DDIM   1024     // D
#define DMLP   4096     // D_MLP
#define NEXP   8        // E
#define CHUNK  256      // D_MLP chunk staged through LDS (prepped kernel)
#define TM     64       // tokens per workgroup tile (prepped kernel)
#define CHUNKF 128      // chunk for fallback kernel
#define TMF    16       // tokens per workgroup tile (fallback kernel)

#define KT1 (DDIM / 32)   // 32   k-tiles for GEMM1 (W_in rows)
#define NT1 (DMLP / 16)   // 256  n-tiles for GEMM1
#define KT2 (DMLP / 32)   // 128  k-tiles for GEMM2 (W_out rows)
#define NT2 (DDIM / 16)   // 64   n-tiles for GEMM2

typedef __attribute__((ext_vector_type(16))) __bf16 v16bf;
typedef __attribute__((ext_vector_type(8)))  __bf16 v8bf;
typedef __attribute__((ext_vector_type(4)))  __bf16 v4bf;
typedef __attribute__((ext_vector_type(8)))  float  v8f;

// ------------------------------------------------------------------
// A-fragment (16x32 bf16) from LDS, row-major with row stride ld.
// lanes 0-15 (row M=lane) hold K = k..k+7 and k+16..k+23,
// lanes 16-31 (row M=lane-16) hold K = k+8..k+15 and k+24..k+31.
// ------------------------------------------------------------------
__device__ __forceinline__ v16bf load_a_lds(const __bf16* base, int ld, int k, int lane) {
    int row  = lane & 15;
    int koff = (lane >> 4) * 8;
    const __bf16* p = base + row * ld + k + koff;
    v8bf lo = *(const v8bf*)(p);
    v8bf hi = *(const v8bf*)(p + 16);
    return __builtin_shufflevector(lo, hi, 0,1,2,3,4,5,6,7,8,9,10,11,12,13,14,15);
}

// Fragment-ready bf16 B tile fetch: 32 B contiguous per lane (2 x b128).
__device__ __forceinline__ v16bf load_b_prep(const __bf16* tiles, size_t tile_idx, int lane) {
    return *(const v16bf*)(tiles + ((tile_idx * 32 + lane) << 4));
}

// Fallback B-fragment (32x16) gathered from global fp32 weights.
__device__ __forceinline__ v16bf load_b_glb(const float* base, int ldb, int lane) {
    int n    = lane & 15;
    int koff = (lane >> 4) * 8;
    const float* p = base + n;
    v16bf b;
#pragma unroll
    for (int j = 0; j < 8; ++j) {
        b[j]     = (__bf16)p[(size_t)(koff + j) * ldb];
        b[j + 8] = (__bf16)p[(size_t)(16 + koff + j) * ldb];
    }
    return b;
}

// ==================================================================
// Prep: x fp32 -> bf16 row-major copy (one float4 per thread).
// ==================================================================
__global__ __launch_bounds__(256)
void prep_x_kernel(const float* __restrict__ src, __bf16* __restrict__ dst, int total4) {
    for (int i = blockIdx.x * 256 + threadIdx.x; i < total4; i += gridDim.x * 256) {
        float4 v = *(const float4*)(src + (size_t)i * 4);
        v4bf o;
        o[0] = (__bf16)v.x; o[1] = (__bf16)v.y; o[2] = (__bf16)v.z; o[3] = (__bf16)v.w;
        *(v4bf*)(dst + (size_t)i * 4) = o;
    }
}

// ==================================================================
// Prep: weights fp32 [nmats][R][C] -> bf16 fragment-ready tiles:
// dst[(((e*KT + kt)*NT + nt)*32 + lane)*16 + j] = W[e][kt*32 + kmap][nt*16 + (lane&15)]
// ==================================================================
__global__ __launch_bounds__(256)
void prep_w_kernel(const float* __restrict__ src, __bf16* __restrict__ dst,
                   int R, int C, int KT, int NT, int total) {
    for (int id = blockIdx.x * 256 + threadIdx.x; id < total; id += gridDim.x * 256) {
        int lane = id & 31;
        int t    = id >> 5;
        int nt   = t % NT;  t /= NT;
        int kt   = t % KT;
        int e    = t / KT;
        int koff = (lane >> 4) * 8;
        int n    = nt * 16 + (lane & 15);
        const float* s = src + ((size_t)e * R + (size_t)kt * 32) * C + n;
        v16bf v;
#pragma unroll
        for (int j = 0; j < 8; ++j) {
            v[j]     = (__bf16)s[(size_t)(koff + j) * C];
            v[j + 8] = (__bf16)s[(size_t)(16 + koff + j) * C];
        }
        *(v16bf*)(dst + (size_t)id * 16) = v;
    }
}

// ==================================================================
// Gating: one wave32 per token; top-2 softmax; append to expert lists.
// ==================================================================
__global__ __launch_bounds__(256)
void moe_gate_kernel(const float* __restrict__ x, const float* __restrict__ Wg,
                     int* __restrict__ counts, int* __restrict__ tok_list,
                     float* __restrict__ w_list, int ntok) {
    int lane  = threadIdx.x & 31;
    int warp  = threadIdx.x >> 5;
    int token = blockIdx.x * 8 + warp;
    if (token >= ntok) return;

    float acc[NEXP];
#pragma unroll
    for (int e = 0; e < NEXP; ++e) acc[e] = 0.f;

    const float* xr = x + (size_t)token * DDIM;
    for (int d = lane; d < DDIM; d += 32) {
        float xv = xr[d];
        const float* wr = Wg + (size_t)d * NEXP;
#pragma unroll
        for (int e = 0; e < NEXP; ++e) acc[e] += xv * wr[e];
    }
#pragma unroll
    for (int e = 0; e < NEXP; ++e) {
#pragma unroll
        for (int off = 16; off > 0; off >>= 1)
            acc[e] += __shfl_xor(acc[e], off, 32);
    }

    if (lane == 0) {
        float b1 = -1e30f, b2 = -1e30f; int e1 = 0, e2 = 0;
#pragma unroll
        for (int e = 0; e < NEXP; ++e)
            if (acc[e] > b1) { b1 = acc[e]; e1 = e; }
#pragma unroll
        for (int e = 0; e < NEXP; ++e)
            if (e != e1 && acc[e] > b2) { b2 = acc[e]; e2 = e; }
        float t  = expf(b2 - b1);
        float w1 = 1.f / (1.f + t);
        float w2 = t * w1;

        int p1 = atomicAdd(&counts[e1], 1);
        tok_list[e1 * ntok + p1] = token;
        w_list [e1 * ntok + p1] = w1;
        int p2 = atomicAdd(&counts[e2], 1);
        tok_list[e2 * ntok + p2] = token;
        w_list [e2 * ntok + p2] = w2;
    }
}

// ==================================================================
// PREPPED fused expert MLP. grid (E, ntok/64), 1024 threads = 32 waves.
// Dynamic LDS: x tile 128 KB (bf16, async-DMA gathered) + h tile 32 KB.
// GEMM1: wave (rp, c1) computes a 32x16 h tile (B loaded once, 2 WMMAs).
// GEMM2: wave owns 32 output cols x 64 rows; each B fragment feeds 4 WMMAs.
// ==================================================================
__global__ __launch_bounds__(1024)
void moe_expert_prepped(const __bf16* __restrict__ xb,
                        const __bf16* __restrict__ WinP,  const float* __restrict__ b_in,
                        const __bf16* __restrict__ WoutP, const float* __restrict__ b_out,
                        const int* __restrict__ counts,   const int* __restrict__ tok_list,
                        const float* __restrict__ w_list, float* __restrict__ out, int ntok) {
    extern __shared__ __attribute__((aligned(16))) char smem[];
    __bf16* x_lds = (__bf16*)smem;                         // 64*1024*2 = 128 KB
    __bf16* h_lds = (__bf16*)(smem + TM * DDIM * 2);       // 64*256*2  =  32 KB
    int*    tid_s = (int*)(smem + TM * DDIM * 2 + TM * CHUNK * 2);
    float*  w_s   = (float*)(smem + TM * DDIM * 2 + TM * CHUNK * 2 + TM * 4);

    const int e    = blockIdx.x;
    const int base = blockIdx.y * TM;
    const int cnt  = counts[e];
    if (base >= cnt) return;
    const int nrows = min(TM, cnt - base);

    const int tid  = threadIdx.x;
    const int lane = tid & 31;
    const int wid  = tid >> 5;          // 0..31

    if (tid < TM) {
        if (tid < nrows) {
            tid_s[tid] = tok_list[e * ntok + base + tid];
            w_s[tid]   = w_list [e * ntok + base + tid];
        } else {
            tid_s[tid] = 0;
            w_s[tid]   = 0.f;
        }
    }
    __syncthreads();

    // ---- async DMA: gather 64 bf16 token rows straight into LDS ----
    // 64 rows x 2048 B = 128 KB; 8 x b128 per thread, ASYNCcnt-tracked.
#pragma unroll
    for (int i = 0; i < 8; ++i) {
        int lin = i * 1024 + tid;         // 16-byte chunk index, 0..8191
        int row = lin >> 7;               // 128 chunks per row
        int off = (lin & 127) * 16;       // byte offset within row
        unsigned lds_off = (unsigned)(uintptr_t)((const char*)x_lds + lin * 16);
        const char* g = (const char*)(xb + (size_t)tid_s[row] * DDIM) + off;
        asm volatile("global_load_async_to_lds_b128 %0, %1, off"
                     :: "v"(lds_off), "v"(g) : "memory");
    }
    asm volatile("s_wait_asynccnt 0x0" ::: "memory");
    __syncthreads();

    const __bf16* WiP = WinP  + (size_t)e * KT1 * NT1 * 512;
    const __bf16* WoP = WoutP + (size_t)e * KT2 * NT2 * 512;
    const float*  bi  = b_in  + (size_t)e * DMLP;
    const float*  bo  = b_out + (size_t)e * DDIM;

    const v8f vzero = {0.f,0.f,0.f,0.f,0.f,0.f,0.f,0.f};
    v8f acc[4][2];                      // [row-tile 0..3][col-tile 0..1], cols wid*32..+32
#pragma unroll
    for (int r = 0; r < 4; ++r)
#pragma unroll
        for (int t = 0; t < 2; ++t) acc[r][t] = vzero;

    const int c1 = wid & 15;            // GEMM1 col-tile (0..15) within chunk
    const int rp = wid >> 4;            // GEMM1 row-pair: rows rp*32 .. rp*32+31
    const int m0 = (lane >> 4) * 8;     // C-layout row base for this lane
    const int nn = lane & 15;           // C-layout column for this lane

    for (int mc = 0; mc < DMLP; mc += CHUNK) {
        // ---- GEMM1: 32x16 h tile at rows rp*32, cols mc + c1*16 (B shared x2) ----
        v8f h0 = vzero, h1 = vzero;
        const int ncol = mc + c1 * 16;
        const int ntg  = (mc >> 4) + c1;
        const __bf16* xrow0 = x_lds + (rp * 32)      * DDIM;
        const __bf16* xrow1 = x_lds + (rp * 32 + 16) * DDIM;
        for (int k = 0; k < DDIM; k += 32) {
            v16bf b  = load_b_prep(WiP, (size_t)(k >> 5) * NT1 + ntg, lane);
            v16bf a0 = load_a_lds(xrow0, DDIM, k, lane);
            v16bf a1 = load_a_lds(xrow1, DDIM, k, lane);
            h0 = __builtin_amdgcn_wmma_f32_16x16x32_bf16(false, a0, false, b,
                                                         (short)0, h0, false, false);
            h1 = __builtin_amdgcn_wmma_f32_16x16x32_bf16(false, a1, false, b,
                                                         (short)0, h1, false, false);
        }
        // bias + exact GELU, write both 16x16 tiles into h_lds
        {
            float bias = bi[ncol + nn];
#pragma unroll
            for (int i = 0; i < 8; ++i) {
                float v = h0[i] + bias;
                v = 0.5f * v * (1.f + erff(v * 0.70710678118f));
                h_lds[(rp * 32 + m0 + i) * CHUNK + c1 * 16 + nn] = (__bf16)v;
                float u = h1[i] + bias;
                u = 0.5f * u * (1.f + erff(u * 0.70710678118f));
                h_lds[(rp * 32 + 16 + m0 + i) * CHUNK + c1 * 16 + nn] = (__bf16)u;
            }
        }
        __syncthreads();   // h_lds fully written before GEMM2 reads

        // ---- GEMM2: acc += h_chunk @ W_out[mc..mc+256, wave cols] ----
        for (int kk = 0; kk < CHUNK; kk += 32) {
            const int kt2 = (mc + kk) >> 5;
            v16bf b0 = load_b_prep(WoP, (size_t)kt2 * NT2 + (wid * 2 + 0), lane);
            v16bf b1 = load_b_prep(WoP, (size_t)kt2 * NT2 + (wid * 2 + 1), lane);
#pragma unroll
            for (int r = 0; r < 4; ++r) {
                v16bf a = load_a_lds(h_lds + r * 16 * CHUNK, CHUNK, kk, lane);
                acc[r][0] = __builtin_amdgcn_wmma_f32_16x16x32_bf16(false, a, false, b0,
                                                                    (short)0, acc[r][0], false, false);
                acc[r][1] = __builtin_amdgcn_wmma_f32_16x16x32_bf16(false, a, false, b1,
                                                                    (short)0, acc[r][1], false, false);
            }
        }
        __syncthreads();   // done with h_lds before next chunk overwrites it
    }

    // ---------- epilogue: + b_out, * routing weight, scatter-add ----------
#pragma unroll
    for (int t = 0; t < 2; ++t) {
        int col = wid * 32 + t * 16 + nn;
        float bov = bo[col];
#pragma unroll
        for (int r = 0; r < 4; ++r) {
#pragma unroll
            for (int i = 0; i < 8; ++i) {
                int m = r * 16 + m0 + i;
                if (m < nrows) {
                    float val = (acc[r][t][i] + bov) * w_s[m];
                    atomicAdd(&out[(size_t)tid_s[m] * DDIM + col], val);
                }
            }
        }
    }
}

// ==================================================================
// FALLBACK fused expert MLP (no workspace weights): per-lane fp32
// gather + convert. grid (E, ntok/16), 256 threads = 8 waves.
// ==================================================================
__global__ __launch_bounds__(256)
void moe_expert_fallback(const float* __restrict__ x,
                         const float* __restrict__ W_in,  const float* __restrict__ b_in,
                         const float* __restrict__ W_out, const float* __restrict__ b_out,
                         const int* __restrict__ counts,  const int* __restrict__ tok_list,
                         const float* __restrict__ w_list, float* __restrict__ out, int ntok) {
    __shared__ __attribute__((aligned(16))) __bf16 x_lds[TMF * DDIM];
    __shared__ __attribute__((aligned(16))) __bf16 h_lds[TMF * CHUNKF];
    __shared__ int   tid_s[TMF];
    __shared__ float w_s[TMF];

    const int e    = blockIdx.x;
    const int base = blockIdx.y * TMF;
    const int cnt  = counts[e];
    if (base >= cnt) return;
    const int nrows = min(TMF, cnt - base);

    const int tid  = threadIdx.x;
    const int lane = tid & 31;
    const int wid  = tid >> 5;

    if (tid < TMF) {
        if (tid < nrows) {
            tid_s[tid] = tok_list[e * ntok + base + tid];
            w_s[tid]   = w_list [e * ntok + base + tid];
        } else {
            tid_s[tid] = 0;
            w_s[tid]   = 0.f;
        }
    }
    __syncthreads();

    for (int lin = tid; lin < TMF * (DDIM / 4); lin += 256) {
        int row = lin >> 8;
        int c4  = lin & 255;
        float4 v = make_float4(0.f, 0.f, 0.f, 0.f);
        if (row < nrows)
            v = *(const float4*)(x + (size_t)tid_s[row] * DDIM + c4 * 4);
        v4bf o;
        o[0] = (__bf16)v.x; o[1] = (__bf16)v.y; o[2] = (__bf16)v.z; o[3] = (__bf16)v.w;
        *(v4bf*)(&x_lds[row * DDIM + c4 * 4]) = o;
    }
    __syncthreads();

    const float* Wi = W_in  + (size_t)e * DDIM * DMLP;
    const float* Wo = W_out + (size_t)e * DMLP * DDIM;
    const float* bi = b_in  + (size_t)e * DMLP;
    const float* bo = b_out + (size_t)e * DDIM;

    const v8f vzero = {0.f,0.f,0.f,0.f,0.f,0.f,0.f,0.f};
    v8f acc[8];
#pragma unroll
    for (int t = 0; t < 8; ++t) acc[t] = vzero;

    const int m0 = (lane >> 4) * 8;
    const int nn = lane & 15;

    for (int mc = 0; mc < DMLP; mc += CHUNKF) {
        v8f h = vzero;
        const int ncol = mc + wid * 16;
        for (int k = 0; k < DDIM; k += 32) {
            v16bf a = load_a_lds(x_lds, DDIM, k, lane);
            v16bf b = load_b_glb(Wi + (size_t)k * DMLP + ncol, DMLP, lane);
            h = __builtin_amdgcn_wmma_f32_16x16x32_bf16(false, a, false, b,
                                                        (short)0, h, false, false);
        }
        {
            float bias = bi[ncol + nn];
#pragma unroll
            for (int i = 0; i < 8; ++i) {
                float v = h[i] + bias;
                v = 0.5f * v * (1.f + erff(v * 0.70710678118f));
                h_lds[(m0 + i) * CHUNKF + wid * 16 + nn] = (__bf16)v;
            }
        }
        __syncthreads();

        for (int kk = 0; kk < CHUNKF; kk += 32) {
            v16bf a = load_a_lds(h_lds, CHUNKF, kk, lane);
#pragma unroll
            for (int t = 0; t < 8; ++t) {
                v16bf b = load_b_glb(Wo + (size_t)(mc + kk) * DDIM + wid * 128 + t * 16, DDIM, lane);
                acc[t] = __builtin_amdgcn_wmma_f32_16x16x32_bf16(false, a, false, b,
                                                                 (short)0, acc[t], false, false);
            }
        }
        __syncthreads();
    }

#pragma unroll
    for (int t = 0; t < 8; ++t) {
        int col = wid * 128 + t * 16 + nn;
        float bov = bo[col];
#pragma unroll
        for (int i = 0; i < 8; ++i) {
            int m = m0 + i;
            if (m < nrows) {
                float val = (acc[t][i] + bov) * w_s[m];
                atomicAdd(&out[(size_t)tid_s[m] * DDIM + col], val);
            }
        }
    }
}

// ==================================================================
extern "C" void kernel_launch(void* const* d_in, const int* in_sizes, int n_in,
                              void* d_out, int out_size, void* d_ws, size_t ws_size,
                              hipStream_t stream) {
    const float* x    = (const float*)d_in[0];
    const float* Wg   = (const float*)d_in[1];
    const float* W_in = (const float*)d_in[2];
    const float* b_in = (const float*)d_in[3];
    const float* W_out= (const float*)d_in[4];
    const float* b_out= (const float*)d_in[5];
    float* out        = (float*)d_out;

    const int ntok = in_sizes[0] / DDIM;   // B*P = 4096

    // ---- workspace layout ----
    size_t off = 0;
    char* ws = (char*)d_ws;
    int*   counts   = (int*)(ws + off);  off += 256;
    int*   tok_list = (int*)(ws + off);  off += (size_t)NEXP * ntok * sizeof(int);
    float* w_list   = (float*)(ws + off); off += (size_t)NEXP * ntok * sizeof(float);
    off = (off + 255) & ~(size_t)255;
    __bf16* xb   = (__bf16*)(ws + off);  off += (size_t)ntok * DDIM * 2;
    __bf16* WinP = (__bf16*)(ws + off);  off += (size_t)NEXP * DDIM * DMLP * 2;
    __bf16* WoutP= (__bf16*)(ws + off);  off += (size_t)NEXP * DMLP * DDIM * 2;
    const int prepped = (ws_size >= off) ? 1 : 0;

    hipMemsetAsync(counts, 0, 256, stream);
    hipMemsetAsync(out, 0, (size_t)out_size * sizeof(float), stream);

    moe_gate_kernel<<<(ntok + 7) / 8, 256, 0, stream>>>(x, Wg, counts, tok_list, w_list, ntok);

    if (prepped) {
        int total4 = ntok * (DDIM / 4);
        prep_x_kernel<<<(total4 + 255) / 256, 256, 0, stream>>>(x, xb, total4);
        int tot1 = NEXP * KT1 * NT1 * 32;
        prep_w_kernel<<<(tot1 + 255) / 256, 256, 0, stream>>>(W_in, WinP, DDIM, DMLP, KT1, NT1, tot1);
        int tot2 = NEXP * KT2 * NT2 * 32;
        prep_w_kernel<<<(tot2 + 255) / 256, 256, 0, stream>>>(W_out, WoutP, DMLP, DDIM, KT2, NT2, tot2);

        const int smem_bytes = TM * DDIM * 2 + TM * CHUNK * 2 + TM * 4 + TM * 4; // 164352
        hipFuncSetAttribute((const void*)moe_expert_prepped,
                            hipFuncAttributeMaxDynamicSharedMemorySize, smem_bytes);
        dim3 grid(NEXP, (ntok + TM - 1) / TM);
        moe_expert_prepped<<<grid, 1024, smem_bytes, stream>>>(xb, WinP, b_in, WoutP, b_out,
                                                               counts, tok_list, w_list, out, ntok);
    } else {
        dim3 grid(NEXP, (ntok + TMF - 1) / TMF);
        moe_expert_fallback<<<grid, 256, 0, stream>>>(x, W_in, b_in, W_out, b_out,
                                                      counts, tok_list, w_list, out, ntok);
    }
}
